// MTSLTM_82025285419106
// MI455X (gfx1250) — matrix operand order repaired and checked by reading, not compiled, not run on hardware
//
#include <hip/hip_runtime.h>
#include <hip/hip_bf16.h>
#include <math.h>

// Problem dims (fixed by the reference)
#define B_   64
#define T_   512
#define D_   256
#define H_   512
#define G4_  2048   // 4*H

typedef __attribute__((ext_vector_type(16))) __bf16       v16bf;
typedef __attribute__((ext_vector_type(8)))  float        v8f;
typedef __attribute__((ext_vector_type(4)))  unsigned int u32x4;

union Frag { u32x4 q[2]; v16bf v; };

__device__ __forceinline__ unsigned short f32_to_bf16(float f) {
  unsigned int x = __float_as_uint(f);
  unsigned int r = x + 0x7FFFu + ((x >> 16) & 1u);   // round-to-nearest-even
  return (unsigned short)(r >> 16);
}

// ---------------- setup kernels ----------------

__global__ void cvt_f32_bf16(const float* __restrict__ src,
                             unsigned short* __restrict__ dst, int n) {
  int i = blockIdx.x * blockDim.x + threadIdx.x;
  int stride = gridDim.x * blockDim.x;
  for (; i < n; i += stride) dst[i] = f32_to_bf16(src[i]);
}

// Pack a row-major f32 [K,N] weight into the exact WMMA B-fragment layout:
// tile (ntile,ktile) of 32x16 -> 1024B; lane l owns 32B = 8 dwords (VGPR v=0..7),
// dword v = bf16 pair B[kb2+2v', col], B[kb2+2v'+1, col], kb2 = 8*(l>=16) + 16*(v>=4), v' = v&3, col = l&15.
__global__ void pack_b(const float* __restrict__ src,
                       unsigned int* __restrict__ dst, int K, int N) {
  int total  = (K * N) >> 1;
  int ktiles = K >> 5;
  int i = blockIdx.x * blockDim.x + threadIdx.x;
  int stride = gridDim.x * blockDim.x;
  for (; i < total; i += stride) {
    int v    = i & 7;
    int lane = (i >> 3) & 31;
    int tile = i >> 8;
    int kt   = tile % ktiles;
    int ntl  = tile / ktiles;
    int klocal = ((lane >> 4) << 3) + ((v & 4) << 2) + ((v & 3) << 1);
    int col    = (ntl << 4) + (lane & 15);
    int k      = (kt << 5) + klocal;
    unsigned int lo = f32_to_bf16(src[(size_t)k * N + col]);
    unsigned int hi = f32_to_bf16(src[(size_t)(k + 1) * N + col]);
    dst[i] = lo | (hi << 16);
  }
}

__global__ void zero_state(unsigned int* __restrict__ hbf,
                           unsigned int* __restrict__ cbf,
                           float* __restrict__ cf) {
  int i = blockIdx.x * blockDim.x + threadIdx.x;      // 32768 threads
  if (i < (B_ * H_ / 2)) { hbf[i] = 0u; cbf[i] = 0u; }
  if (i < (B_ * H_))     { cf[i] = 0.0f; }
}

// ---------------- fused recurrent step ----------------
// grid: 32 blocks (one 16-wide column tile of H each), 128 threads (4 waves).
// Wave w owns B-rows [16w,16w+16). Each wave accumulates 5 WMMA tile chains:
//   c_s   = c_t @ W_d[:,tile]                         (K=512)
//   gate_g = x_t @ W[:,g*H+tile] + h_t @ U[:,g*H+tile] (K=256 + K=512), g=0..3
__global__ __launch_bounds__(128) void tlstm_step(
    const unsigned short* __restrict__ x_bf,   // [B,T,D] bf16
    const unsigned short* __restrict__ Wp,     // packed K=256,N=2048
    const unsigned short* __restrict__ Up,     // packed K=512,N=2048
    const unsigned short* __restrict__ Wdp,    // packed K=512,N=512
    const unsigned short* __restrict__ h_src,  // [B,H] bf16
    const unsigned short* __restrict__ c_src,  // [B,H] bf16
    const float* __restrict__ cf_src,          // [B,H] f32
    const float* __restrict__ td,              // [B,T]
    const float* __restrict__ bias,            // [4H]
    unsigned short* __restrict__ h_dst,
    unsigned short* __restrict__ c_dst,
    float* __restrict__ cf_dst,
    float* __restrict__ out,                   // [B,T,H] ++ [B,H] ++ [B,H]
    int t) {
  const int lane = threadIdx.x & 31;
  const int wave = threadIdx.x >> 5;
  const int nt   = blockIdx.x;            // column tile of H (16 cols)
  const int row0 = wave << 4;             // base B-row of this wave's tiles
  const int aRow = row0 + (lane & 15);    // A-fragment row for this lane
  const int kbA  = (lane >> 4) << 3;      // 0 or 8 (A-layout k base)
  const int col  = (nt << 4) + (lane & 15);

  v8f acc_cs = {};
  v8f accg[4] = {{}, {}, {}, {}};

  // ---- x_t @ W  (K = 256) ----
  {
    const unsigned short* arow = x_bf + ((size_t)aRow * T_ + t) * D_;
    for (int kt = 0; kt < D_ / 32; ++kt) {
      Frag a;
      a.q[0] = *(const u32x4*)(arow + kt * 32 + kbA);
      a.q[1] = *(const u32x4*)(arow + kt * 32 + kbA + 16);
#pragma unroll
      for (int g = 0; g < 4; ++g) {
        const unsigned short* bt =
            Wp + ((size_t)((g * 32 + nt) * (D_ / 32) + kt) << 9) + (lane << 4);
        Frag b;
        b.q[0] = *(const u32x4*)bt;
        b.q[1] = *(const u32x4*)(bt + 8);
        accg[g] = __builtin_amdgcn_wmma_f32_16x16x32_bf16(
            false, a.v, false, b.v, (short)0, accg[g], false, false);
      }
    }
  }
  // ---- h_t @ U  (K = 512) ----
  {
    const unsigned short* arow = h_src + (size_t)aRow * H_;
    for (int kt = 0; kt < H_ / 32; ++kt) {
      Frag a;
      a.q[0] = *(const u32x4*)(arow + kt * 32 + kbA);
      a.q[1] = *(const u32x4*)(arow + kt * 32 + kbA + 16);
#pragma unroll
      for (int g = 0; g < 4; ++g) {
        const unsigned short* bt =
            Up + ((size_t)((g * 32 + nt) * (H_ / 32) + kt) << 9) + (lane << 4);
        Frag b;
        b.q[0] = *(const u32x4*)bt;
        b.q[1] = *(const u32x4*)(bt + 8);
        accg[g] = __builtin_amdgcn_wmma_f32_16x16x32_bf16(
            false, a.v, false, b.v, (short)0, accg[g], false, false);
      }
    }
  }
  // ---- c_t @ W_d  (K = 512) ----
  {
    const unsigned short* arow = c_src + (size_t)aRow * H_;
    for (int kt = 0; kt < H_ / 32; ++kt) {
      Frag a;
      a.q[0] = *(const u32x4*)(arow + kt * 32 + kbA);
      a.q[1] = *(const u32x4*)(arow + kt * 32 + kbA + 16);
      const unsigned short* bt =
          Wdp + ((size_t)(nt * (H_ / 32) + kt) << 9) + (lane << 4);
      Frag b;
      b.q[0] = *(const u32x4*)bt;
      b.q[1] = *(const u32x4*)(bt + 8);
      acc_cs = __builtin_amdgcn_wmma_f32_16x16x32_bf16(
          false, a.v, false, b.v, (short)0, acc_cs, false, false);
    }
  }

  // ---- elementwise LSTM cell + time decay ----
  const float bi = bias[col];
  const float bf = bias[H_ + col];
  const float bg = bias[2 * H_ + col];
  const float bo = bias[3 * H_ + col];
  const int rOff = (lane >> 4) << 3;   // C/D layout: rows row0+rOff+r, col = lane&15

#pragma unroll
  for (int r = 0; r < 8; ++r) {
    const int row = row0 + rOff + r;
    const float ct  = cf_src[(size_t)row * H_ + col];
    const float tdv = td[(size_t)row * T_ + t];
    const float inv = 1.0f / logf(2.718281828459045f + tdv);  // 1/log(e+dt)
    const float cs  = tanhf(acc_cs[r]);
    const float cadj = (ct - cs) + cs * inv;
    const float gi = 1.0f / (1.0f + expf(-(accg[0][r] + bi)));
    const float gf = 1.0f / (1.0f + expf(-(accg[1][r] + bf)));
    const float gg = tanhf(accg[2][r] + bg);
    const float go = 1.0f / (1.0f + expf(-(accg[3][r] + bo)));
    const float cn = gf * cadj + gi * gg;
    const float tc = tanhf(cn);          // c_out = tanh(c_new)
    const float ho = tanhf(go * tc);     // h_out = tanh(o * tanh(c_new))

    out[((size_t)row * T_ + t) * H_ + col] = ho;     // hidden_seq[b,t,:]
    h_dst[(size_t)row * H_ + col]  = f32_to_bf16(ho);
    c_dst[(size_t)row * H_ + col]  = f32_to_bf16(tc);
    cf_dst[(size_t)row * H_ + col] = tc;
    if (t == T_ - 1) {
      out[(size_t)B_ * T_ * H_ + (size_t)row * H_ + col] = ho;             // h_t
      out[(size_t)B_ * T_ * H_ + (size_t)B_ * H_ + (size_t)row * H_ + col] = tc; // c_t
    }
  }
}

// ---------------- host side ----------------

extern "C" void kernel_launch(void* const* d_in, const int* in_sizes, int n_in,
                              void* d_out, int out_size, void* d_ws, size_t ws_size,
                              hipStream_t stream) {
  const float* inputs = (const float*)d_in[0];  // [B,T,D]
  const float* tdel   = (const float*)d_in[1];  // [B,T]
  const float* W_d    = (const float*)d_in[2];  // [H,H]
  const float* W      = (const float*)d_in[3];  // [D,4H]
  const float* U      = (const float*)d_in[4];  // [H,4H]
  const float* bias   = (const float*)d_in[5];  // [4H]
  float* out = (float*)d_out;

  char* ws = (char*)d_ws;
  size_t off = 0;
  auto alloc = [&](size_t bytes) -> char* {
    char* p = ws + off;
    off += (bytes + 255) & ~(size_t)255;
    return p;
  };
  unsigned short* x_bf = (unsigned short*)alloc((size_t)B_ * T_ * D_ * 2);
  unsigned short* Wp   = (unsigned short*)alloc((size_t)D_ * G4_ * 2);
  unsigned short* Up   = (unsigned short*)alloc((size_t)H_ * G4_ * 2);
  unsigned short* Wdp  = (unsigned short*)alloc((size_t)H_ * H_ * 2);
  unsigned short* hbf[2], * cbf[2];
  float* cf[2];
  for (int i = 0; i < 2; ++i) {
    hbf[i] = (unsigned short*)alloc((size_t)B_ * H_ * 2);
    cbf[i] = (unsigned short*)alloc((size_t)B_ * H_ * 2);
    cf[i]  = (float*)alloc((size_t)B_ * H_ * 4);
  }

  // 1) convert activations to bf16; pack weights into WMMA B-fragment layout
  cvt_f32_bf16<<<4096, 256, 0, stream>>>(inputs, x_bf, B_ * T_ * D_);
  pack_b<<<(D_ * G4_ / 2 + 255) / 256, 256, 0, stream>>>(W,   (unsigned int*)Wp,  D_, G4_);
  pack_b<<<(H_ * G4_ / 2 + 255) / 256, 256, 0, stream>>>(U,   (unsigned int*)Up,  H_, G4_);
  pack_b<<<(H_ * H_  / 2 + 255) / 256, 256, 0, stream>>>(W_d, (unsigned int*)Wdp, H_, H_);
  zero_state<<<(B_ * H_ + 255) / 256, 256, 0, stream>>>(
      (unsigned int*)hbf[0], (unsigned int*)cbf[0], cf[0]);

  // 2) sequential recurrence: one fused kernel per timestep, ping-pong carries
  for (int t = 0; t < T_; ++t) {
    const int s = t & 1, d = s ^ 1;
    tlstm_step<<<H_ / 16, 128, 0, stream>>>(
        x_bf, Wp, Up, Wdp,
        hbf[s], cbf[s], cf[s],
        tdel, bias,
        hbf[d], cbf[d], cf[d],
        out, t);
  }
}
